// AdvancedGraphormer_22016002359714
// MI455X (gfx1250) — compile-verified, matrix operand order checked by reading, not run on hardware
//
#include <hip/hip_runtime.h>
#include <math.h>

typedef __attribute__((ext_vector_type(16))) __bf16 v16bf;
typedef __attribute__((ext_vector_type(8)))  float  v8f;

#define DEVINL static __device__ __forceinline__

constexpr int   cB  = 128;
constexpr int   cNT = 16384;
constexpr int   cE  = 131072;
constexpr float cSCALE = 0.17677669529663687f; // 1/sqrt(32)

DEVINL __bf16 f2bf(float f) {
  union { float f; unsigned u; } x; x.f = f;
  unsigned r = (x.u + 0x7FFFu + ((x.u >> 16) & 1u)) >> 16;
  union { unsigned short s; __bf16 h; } y; y.s = (unsigned short)r;
  return y.h;
}

template<int ACT> DEVINL float actf(float v) {
  if (ACT == 1) return v > 0.f ? v : 0.f;
  if (ACT == 2) return 0.5f * v * (1.f + erff(v * 0.7071067811865475f));
  if (ACT == 3) return 1.f / (1.f + expf(-v));
  return v;
}

DEVINL float atomicMaxF(float* addr, float val) {
  int* ia = (int*)addr;
  int old = *ia;
  while (__int_as_float(old) < val) {
    int assumed = old;
    old = atomicCAS(ia, assumed, __float_as_int(val));
    if (old == assumed) break;
  }
  return __int_as_float(old);
}

// ---------------- weight fp32 -> WMMA-fragment-packed bf16 ----------------
// Packed layout: for tile (kt, nt): 32 lanes x 16 bf16 contiguous, i.e.
// P[((kt*ntiles + nt)*32 + lane)*16 + e], where for lane (kh=lane>>4, ln=lane&15):
//   k = kt*32 + (e<8 ? kh*8+e : 16+kh*8+(e-8)),  n = nt*16 + ln
// This matches the 16-bit B-operand VGPR layout, so a lane's whole fragment is
// one 32-byte contiguous load in the GEMM.
__global__ void k_pack_w(const float* __restrict__ W, __bf16* __restrict__ P,
                         int K, int N, int ntiles) {
  int t = blockIdx.x * 256 + threadIdx.x;
  int total = (K >> 5) * ntiles * 512;
  if (t >= total) return;
  int e    = t & 15;
  int lane = (t >> 4) & 31;
  int tile = t >> 9;
  int nt = tile % ntiles, kt = tile / ntiles;
  int kh = lane >> 4, ln = lane & 15;
  int k = kt * 32 + ((e < 8) ? (kh * 8 + e) : (16 + kh * 8 + (e - 8)));
  int n = nt * 16 + ln;
  P[t] = (n < N) ? f2bf(W[(size_t)k * N + n]) : f2bf(0.f);
}

// ---------------- GEMM: C[M,N] = act(A[M,K]*W[K,N] + bias[N]) ----------------
// 256 threads = 8 waves; block tile 64x128; wave tile 16x64 (4 WMMA accs, one
// shared A fragment per k-step). A: 4x b128 loads; B: 4x v16bf (2x b128 each).
template<int ACT>
__global__ __launch_bounds__(256) void k_gemm(
    const float* __restrict__ A, const __bf16* __restrict__ Wp,
    const float* __restrict__ bias, float* __restrict__ C,
    int M, int K, int N)
{
  const int lane = threadIdx.x & 31;
  const int wave = threadIdx.x >> 5;
  const int kh = lane >> 4;
  const int ln = lane & 15;
  const int row0 = blockIdx.y * 64 + (wave >> 1) * 16;
  const int col0 = blockIdx.x * 128 + (wave & 1) * 64;
  const int ntiles = (N + 15) >> 4;
  const int nt0 = col0 >> 4;
  const int mA = row0 + ln;
  v8f acc0 = {}, acc1 = {}, acc2 = {}, acc3 = {};
  const int ktiles = K >> 5;
  for (int kt = 0; kt < ktiles; ++kt) {
    const int k0 = kt << 5;
    v16bf a;
    if (mA < M) {
      const float4* p = (const float4*)(A + (size_t)mA * K + k0 + kh * 8);
      float4 r0 = p[0], r1 = p[1], r2 = p[4], r3 = p[5];
      a[0] = f2bf(r0.x); a[1] = f2bf(r0.y); a[2]  = f2bf(r0.z); a[3]  = f2bf(r0.w);
      a[4] = f2bf(r1.x); a[5] = f2bf(r1.y); a[6]  = f2bf(r1.z); a[7]  = f2bf(r1.w);
      a[8] = f2bf(r2.x); a[9] = f2bf(r2.y); a[10] = f2bf(r2.z); a[11] = f2bf(r2.w);
      a[12] = f2bf(r3.x); a[13] = f2bf(r3.y); a[14] = f2bf(r3.z); a[15] = f2bf(r3.w);
      if (kt + 1 < ktiles) __builtin_prefetch(A + (size_t)mA * K + k0 + 32, 0, 3);
    } else {
      const __bf16 zb = f2bf(0.f);
#pragma unroll
      for (int j = 0; j < 16; ++j) a[j] = zb;
    }
    const __bf16* base = Wp + ((size_t)kt * ntiles + nt0) * 512 + lane * 16;
    if (nt0 + 0 < ntiles) { v16bf b = *(const v16bf*)(base);
      acc0 = __builtin_amdgcn_wmma_f32_16x16x32_bf16(false, a, false, b, (short)0, acc0, false, false); }
    if (nt0 + 1 < ntiles) { v16bf b = *(const v16bf*)(base + 512);
      acc1 = __builtin_amdgcn_wmma_f32_16x16x32_bf16(false, a, false, b, (short)0, acc1, false, false); }
    if (nt0 + 2 < ntiles) { v16bf b = *(const v16bf*)(base + 1024);
      acc2 = __builtin_amdgcn_wmma_f32_16x16x32_bf16(false, a, false, b, (short)0, acc2, false, false); }
    if (nt0 + 3 < ntiles) { v16bf b = *(const v16bf*)(base + 1536);
      acc3 = __builtin_amdgcn_wmma_f32_16x16x32_bf16(false, a, false, b, (short)0, acc3, false, false); }
  }
#pragma unroll
  for (int r = 0; r < 8; ++r) {
    int m = row0 + r + 8 * kh;
    if (m >= M) continue;
    float* crow = C + (size_t)m * N;
    int n0 = col0 + ln;
    if (n0      < N) crow[n0]      = actf<ACT>(acc0[r] + bias[n0]);
    if (n0 + 16 < N) crow[n0 + 16] = actf<ACT>(acc1[r] + bias[n0 + 16]);
    if (n0 + 32 < N) crow[n0 + 32] = actf<ACT>(acc2[r] + bias[n0 + 32]);
    if (n0 + 48 < N) crow[n0 + 48] = actf<ACT>(acc3[r] + bias[n0 + 48]);
  }
}

// ---------------- elementwise helpers ----------------
__global__ void k_fill_f(float* p, float v, long n) {
  long i = (long)blockIdx.x * 256 + threadIdx.x; if (i < n) p[i] = v;
}
__global__ void k_fill_i(int* p, int v, long n) {
  long i = (long)blockIdx.x * 256 + threadIdx.x; if (i < n) p[i] = v;
}
__global__ void k_copy(float* d, const float* s, long n) {
  long i = (long)blockIdx.x * 256 + threadIdx.x; if (i < n) d[i] = s[i];
}
__global__ void k_add2(const float* a, const float* b, float* o, long n) {
  long i = (long)blockIdx.x * 256 + threadIdx.x; if (i < n) o[i] = a[i] + b[i];
}
__global__ void k_add3(const float* a, const float* b, const float* c, float* o, long n) {
  long i = (long)blockIdx.x * 256 + threadIdx.x; if (i < n) o[i] = a[i] + b[i] + c[i];
}
__global__ void k_bn(const float* in, float* o, const float* g, const float* bb,
                     const float* m, const float* v, long n, int C) {
  long i = (long)blockIdx.x * 256 + threadIdx.x; if (i >= n) return;
  int c = (int)(i % C);
  o[i] = (in[i] - m[c]) * rsqrtf(v[c] + 1e-5f) * g[c] + bb[c];
}
__global__ void k_ln(const float* in, float* o, const float* g, const float* bb,
                     int rows, int C) {
  int row = blockIdx.x * 8 + (threadIdx.x >> 5);
  int lane = threadIdx.x & 31;
  if (row >= rows) return;
  const float* p = in + (size_t)row * C;
  float s = 0.f, ss = 0.f;
  for (int c = lane; c < C; c += 32) { float v = p[c]; s += v; ss += v * v; }
  for (int m = 16; m >= 1; m >>= 1) { s += __shfl_xor(s, m, 32); ss += __shfl_xor(ss, m, 32); }
  float mean = s / C, var = ss / C - mean * mean;
  float inv = rsqrtf(var + 1e-5f);
  float* q = o + (size_t)row * C;
  for (int c = lane; c < C; c += 32) q[c] = (p[c] - mean) * inv * g[c] + bb[c];
}
__global__ void k_gather_emb(const int* idx, const float* tab, float* o,
                             int rows, int D, int mxv) {
  long t = (long)blockIdx.x * 256 + threadIdx.x;
  if (t >= (long)rows * D) return;
  int r = (int)(t / D), c = (int)(t - (long)r * D);
  int id = idx[r]; id = id < 0 ? 0 : (id >= mxv ? mxv - 1 : id);
  o[t] = tab[(size_t)id * D + c];
}
__global__ void k_add_gather(const float* x, const int* idx, const float* tab, float* o,
                             int rows, int D, int mxv) {
  long t = (long)blockIdx.x * 256 + threadIdx.x;
  if (t >= (long)rows * D) return;
  int r = (int)(t / D), c = (int)(t - (long)r * D);
  int id = idx[r]; id = id < 0 ? 0 : (id >= mxv ? mxv - 1 : id);
  o[t] = x[t] + tab[(size_t)id * D + c];
}
__global__ void k_concat2(const float* a, int Da, const float* b, int Db, float* o, int rows) {
  int D = Da + Db;
  long t = (long)blockIdx.x * 256 + threadIdx.x;
  if (t >= (long)rows * D) return;
  int r = (int)(t / D), c = (int)(t - (long)r * D);
  o[t] = (c < Da) ? a[(size_t)r * Da + c] : b[(size_t)r * Db + (c - Da)];
}
__global__ void k_fuse(const float* g, const float* a, const float* b, float* o, long n) {
  long i = (long)blockIdx.x * 256 + threadIdx.x; if (i >= n) return;
  float gv = g[i];
  o[i] = gv * a[i] + (1.f - gv) * b[i];
}
__global__ void k_deg(const int* src, int* deg, int E) {
  int e = blockIdx.x * 256 + threadIdx.x;
  if (e < E) atomicAdd(&deg[src[e]], 1);
}

// ---------------- edge attention (TransformerConv) ----------------
__global__ void k_edge_score(const float* q, const float* kv, const float* ee,
                             const int* src, const int* dst, float* S, int E) {
  long t = (long)blockIdx.x * 256 + threadIdx.x;
  if (t >= (long)E * 8) return;
  int e = (int)(t >> 3), h = (int)(t & 7);
  const float* qp = q  + (size_t)dst[e] * 256 + h * 32;
  const float* kp = kv + (size_t)src[e] * 256 + h * 32;
  const float* ep = ee + (size_t)e * 256 + h * 32;
  float a = 0.f;
  for (int i = 0; i < 32; ++i) a += qp[i] * (kp[i] + ep[i]);
  S[t] = a * cSCALE;
}
__global__ void k_seg_max(const float* S, const int* idx, float* mx, int E, int H) {
  long t = (long)blockIdx.x * 256 + threadIdx.x;
  if (t >= (long)E * H) return;
  int e = (int)(t / H), h = (int)(t - (long)e * H);
  atomicMaxF(&mx[(size_t)idx[e] * H + h], S[t]);
}
__global__ void k_seg_expsum(float* S, const int* idx, const float* mx, float* den, int E, int H) {
  long t = (long)blockIdx.x * 256 + threadIdx.x;
  if (t >= (long)E * H) return;
  int e = (int)(t / H), h = (int)(t - (long)e * H);
  size_t g = (size_t)idx[e] * H + h;
  float ex = expf(S[t] - mx[g]);
  S[t] = ex;
  atomicAdd(&den[g], ex);
}
__global__ void k_seg_norm(float* S, const int* idx, const float* den, int E, int H) {
  long t = (long)blockIdx.x * 256 + threadIdx.x;
  if (t >= (long)E * H) return;
  int e = (int)(t / H), h = (int)(t - (long)e * H);
  S[t] = S[t] / (den[(size_t)idx[e] * H + h] + 1e-16f);
}
__global__ void k_edge_agg(const float* v, const float* ee, const float* alpha,
                           const int* src, const int* dst, float* agg, int E) {
  long t = (long)blockIdx.x * 256 + threadIdx.x;
  if (t >= (long)E * 256) return;
  int e = (int)(t >> 8), c = (int)(t & 255);
  int h = c >> 5;
  float val = (v[(size_t)src[e] * 256 + c] + ee[(size_t)e * 256 + c]) * alpha[(size_t)e * 8 + h];
  atomicAdd(&agg[(size_t)dst[e] * 256 + c], val);
}
__global__ void k_eu_concat(const float* xn, const int* src, const int* dst,
                            const float* ea, float* o, int E) {
  long t = (long)blockIdx.x * 256 + threadIdx.x;
  if (t >= (long)E * 640) return;
  int e = (int)(t / 640), c = (int)(t - (long)e * 640);
  float v;
  if (c < 256)      v = xn[(size_t)src[e] * 256 + c];
  else if (c < 512) v = xn[(size_t)dst[e] * 256 + (c - 256)];
  else              v = ea[(size_t)e * 128 + (c - 512)];
  o[t] = v;
}
__global__ void k_eu_update(const float* ea, const float* raw, float* o, int E) {
  long t = (long)blockIdx.x * 256 + threadIdx.x;
  if (t >= (long)E * 128) return;
  int e = (int)(t >> 7), c = (int)(t & 127);
  float delta = raw[(size_t)e * 256 + c];
  float gl    = raw[(size_t)e * 256 + 128 + c];
  o[t] = ea[t] + delta / (1.f + expf(-gl));
}

// ---------------- global attention ----------------
__global__ void k_dm(const int* src, const int* dst, const int* spd, int* dm, int E) {
  int e = blockIdx.x * 256 + threadIdx.x;
  if (e >= E) return;
  int s = src[e];
  int g = s >> 7;
  atomicAdd(&dm[(size_t)g * 16384 + (s & 127) * 128 + (dst[e] & 127)], spd[e]);
}
__global__ void k_bias(const float* pos, const int* dm, const float* db,
                       const float* rw, const float* rb, float* biasT, int total) {
  int t = blockIdx.x * 256 + threadIdx.x;
  if (t >= total) return;
  int b = t >> 14;
  int r = t & 16383;
  int i = r >> 7, j = r & 127;
  const float* pi = pos + (size_t)(b * 128 + i) * 3;
  const float* pj = pos + (size_t)(b * 128 + j) * 3;
  float dx = pi[0] - pj[0], dy = pi[1] - pj[1], dz = pi[2] - pj[2];
  float d = sqrtf(fmaxf(dx * dx + dy * dy + dz * dz, 1e-12f));
  int dv = dm[t]; dv = dv < 0 ? 0 : (dv > 31 ? 31 : dv);
  float acc[8];
#pragma unroll
  for (int h = 0; h < 8; ++h) acc[h] = db[dv * 8 + h] + rb[h];
  const float step = 20.f / 31.f;
  const float coeff = -0.5f / (step * step);
  for (int kx = 0; kx < 32; ++kx) {
    float dd = d - step * kx;
    float ex = expf(coeff * dd * dd);
#pragma unroll
    for (int h = 0; h < 8; ++h) acc[h] += ex * rw[kx * 8 + h];
  }
#pragma unroll
  for (int h = 0; h < 8; ++h) biasT[(size_t)t * 8 + h] = acc[h];
}

// one workgroup per (batch, head): flash-style dense attention with WMMA.
__global__ __launch_bounds__(256) void k_gattn(
    const float* __restrict__ Q, const float* __restrict__ K,
    const float* __restrict__ V, const float* __restrict__ biasT,
    float* __restrict__ O)
{
  int bg = blockIdx.x >> 3;
  int h  = blockIdx.x & 7;
  __shared__ __bf16 Qs[128 * 32];
  __shared__ __bf16 Ks[128 * 32];
  __shared__ __bf16 Vs[128 * 32];
  __shared__ __bf16 Ps[128 * 128];
  int tid = threadIdx.x;
  for (int i = tid; i < 4096; i += 256) {
    int n = i >> 5, d = i & 31;
    size_t off = (size_t)(bg * 128 + n) * 256 + h * 32 + d;
    Qs[i] = f2bf(Q[off]); Ks[i] = f2bf(K[off]); Vs[i] = f2bf(V[off]);
  }
  __syncthreads();
  int lane = tid & 31, wave = tid >> 5;
  int kh = lane >> 4, ln = lane & 15;
  int m0 = wave * 16;
  v16bf a;
  {
    int m = m0 + ln;
#pragma unroll
    for (int j = 0; j < 8; ++j) {
      int k = (j < 4) ? (kh * 8 + 2 * j) : (16 + kh * 8 + 2 * (j - 4));
      a[2 * j] = Qs[m * 32 + k]; a[2 * j + 1] = Qs[m * 32 + k + 1];
    }
  }
  v8f acc[8];
#pragma unroll
  for (int t = 0; t < 8; ++t) {
    v16bf bfrag;
    int n = t * 16 + ln;
#pragma unroll
    for (int j = 0; j < 8; ++j) {
      int k = (j < 4) ? (kh * 8 + 2 * j) : (16 + kh * 8 + 2 * (j - 4));
      bfrag[2 * j] = Ks[n * 32 + k]; bfrag[2 * j + 1] = Ks[n * 32 + k + 1];
    }
    v8f z = {};
    acc[t] = __builtin_amdgcn_wmma_f32_16x16x32_bf16(false, a, false, bfrag, (short)0, z, false, false);
  }
#pragma unroll
  for (int t = 0; t < 8; ++t)
#pragma unroll
    for (int r = 0; r < 8; ++r) {
      int m = m0 + r + 8 * kh;
      int n = t * 16 + ln;
      acc[t][r] = acc[t][r] * cSCALE + biasT[((size_t)(bg * 128 + m) * 128 + n) * 8 + h];
    }
#pragma unroll
  for (int r = 0; r < 8; ++r) {
    float mx = -1e30f;
#pragma unroll
    for (int t = 0; t < 8; ++t) mx = fmaxf(mx, acc[t][r]);
    for (int msk = 8; msk >= 1; msk >>= 1) mx = fmaxf(mx, __shfl_xor(mx, msk, 32));
    float sum = 0.f;
#pragma unroll
    for (int t = 0; t < 8; ++t) { float e = expf(acc[t][r] - mx); acc[t][r] = e; sum += e; }
    for (int msk = 8; msk >= 1; msk >>= 1) sum += __shfl_xor(sum, msk, 32);
    float inv = 1.f / sum;
#pragma unroll
    for (int t = 0; t < 8; ++t) acc[t][r] *= inv;
  }
#pragma unroll
  for (int t = 0; t < 8; ++t)
#pragma unroll
    for (int r = 0; r < 8; ++r) {
      int m = m0 + r + 8 * kh;
      int n = t * 16 + ln;
      Ps[m * 128 + n] = f2bf(acc[t][r]);
    }
  __syncthreads();
  v8f o0 = {}, o1 = {};
  for (int k0 = 0; k0 < 128; k0 += 32) {
    v16bf pa;
    int m = m0 + ln;
#pragma unroll
    for (int j = 0; j < 8; ++j) {
      int k = k0 + ((j < 4) ? (kh * 8 + 2 * j) : (16 + kh * 8 + 2 * (j - 4)));
      pa[2 * j] = Ps[m * 128 + k]; pa[2 * j + 1] = Ps[m * 128 + k + 1];
    }
    v16bf b0, b1;
#pragma unroll
    for (int j = 0; j < 8; ++j) {
      int k = k0 + ((j < 4) ? (kh * 8 + 2 * j) : (16 + kh * 8 + 2 * (j - 4)));
      b0[2 * j]     = Vs[k * 32 + ln];
      b0[2 * j + 1] = Vs[(k + 1) * 32 + ln];
      b1[2 * j]     = Vs[k * 32 + 16 + ln];
      b1[2 * j + 1] = Vs[(k + 1) * 32 + 16 + ln];
    }
    o0 = __builtin_amdgcn_wmma_f32_16x16x32_bf16(false, pa, false, b0, (short)0, o0, false, false);
    o1 = __builtin_amdgcn_wmma_f32_16x16x32_bf16(false, pa, false, b1, (short)0, o1, false, false);
  }
#pragma unroll
  for (int r = 0; r < 8; ++r) {
    int m = m0 + r + 8 * kh;
    size_t off = (size_t)(bg * 128 + m) * 256 + h * 32;
    O[off + ln] = o0[r];
    O[off + 16 + ln] = o1[r];
  }
}

// ---------------- readout helpers ----------------
__global__ void k_x_score(const float* qd, const float* kd, const int* src, float* S, int E) {
  long t = (long)blockIdx.x * 256 + threadIdx.x;
  if (t >= (long)E * 4) return;
  int e = (int)(t >> 2), h = (int)(t & 3);
  const float* qp = qd + (size_t)src[e] * 128 + h * 32;
  const float* kp = kd + (size_t)e * 128 + h * 32;
  float a = 0.f;
  for (int i = 0; i < 32; ++i) a += qp[i] * kp[i];
  S[t] = a * cSCALE;
}
__global__ void k_x_agg(const float* vd, const float* we, const int* src, float* out, int E) {
  long t = (long)blockIdx.x * 256 + threadIdx.x;
  if (t >= (long)E * 128) return;
  int e = (int)(t >> 7), c = (int)(t & 127);
  int h = c >> 5;
  atomicAdd(&out[(size_t)src[e] * 128 + c], vd[t] * we[(size_t)e * 4 + h]);
}
__global__ void k_pool(const float* x, float* meanp, float* maxp) {
  int t = blockIdx.x * 256 + threadIdx.x;
  if (t >= cB * 256) return;
  int b = t >> 8, c = t & 255;
  const float* p = x + (size_t)b * 128 * 256 + c;
  float s = 0.f, m = -1e30f;
  for (int n = 0; n < 128; ++n) { float v = p[(size_t)n * 256]; s += v; m = fmaxf(m, v); }
  meanp[t] = s * (1.f / 128.f);
  maxp[t] = m;
}
__global__ void k_emean(const float* ea, const int* src, float* out, int E) {
  long t = (long)blockIdx.x * 256 + threadIdx.x;
  if (t >= (long)E * 128) return;
  int e = (int)(t >> 7), c = (int)(t & 127);
  int g = src[e] >> 7;
  atomicAdd(&out[(size_t)g * 128 + c], ea[t] * (1.f / 1024.f));
}
__global__ void k_cat4(const float* a, const float* b, const float* gp, const float* ge,
                       const int* batch, float* o, int rows) {
  long t = (long)blockIdx.x * 256 + threadIdx.x;
  if (t >= (long)rows * 512) return;
  int r = (int)(t >> 9), c = (int)(t & 511);
  float v;
  if (c < 128)      v = a[(size_t)r * 128 + c];
  else if (c < 256) v = b[(size_t)r * 128 + (c - 128)];
  else {
    int bb = batch[r];
    if (c < 384) v = gp[(size_t)bb * 128 + (c - 256)];
    else         v = ge[(size_t)bb * 128 + (c - 384)];
  }
  o[t] = v;
}

// ================= host orchestration =================
extern "C" void kernel_launch(void* const* d_in, const int* in_sizes, int n_in,
                              void* d_out, int out_size, void* d_ws, size_t ws_size,
                              hipStream_t stream) {
  (void)in_sizes; (void)n_in; (void)out_size; (void)ws_size;
  auto F  = [&](int i) { return (const float*)d_in[i]; };
  auto IP = [&](int i) { return (const int*)d_in[i]; };
  auto G1 = [](long n) { return dim3((unsigned)((n + 255) / 256)); };

  const float* in_x  = F(0);
  const int*   src   = IP(1);
  const int*   dst   = IP(1) + cE;
  const float* in_ea = F(2);
  const int*   spd   = IP(3);
  const float* pos   = F(4);
  const int*   batch = IP(5);

  // bump allocator over workspace
  char* wsp = (char*)d_ws;
  auto allocB = [&](size_t bytes) { void* p = (void*)wsp; wsp += (bytes + 255) & ~(size_t)255; return p; };
  auto allocF = [&](size_t n) { return (float*)allocB(n * 4); };
  auto allocI = [&](size_t n) { return (int*)allocB(n * 4); };

  __bf16* Wbf = (__bf16*)allocB((size_t)262144 * 2);

  float* x0    = allocF((size_t)cNT * 256);
  float* xcur  = allocF((size_t)cNT * 256);
  float* bufA  = allocF((size_t)cNT * 256);
  float* bufH  = allocF((size_t)cNT * 256);
  float* qb    = allocF((size_t)cNT * 256);
  float* kb    = allocF((size_t)cNT * 256);
  float* vb    = allocF((size_t)cNT * 256);
  float* skp   = allocF((size_t)cNT * 256);
  float* agg   = allocF((size_t)cNT * 256);
  float* xat   = allocF((size_t)cNT * 256);
  float* ffn_t = allocF((size_t)cNT * 1024);
  float* ffn_o = allocF((size_t)cNT * 256);
  float* xnb   = allocF((size_t)cNT * 256);
  float* xcomb = allocF((size_t)cNT * 256);
  float* hN    = allocF((size_t)cNT * 256);
  float* qg    = allocF((size_t)cNT * 256);
  float* kg    = allocF((size_t)cNT * 256);
  float* vg    = allocF((size_t)cNT * 256);
  float* ao    = allocF((size_t)cNT * 256);
  float* xh[4]; for (int l = 0; l < 4; ++l) xh[l] = allocF((size_t)cNT * 256);
  float* ea_hc = allocF((size_t)cE * 128);
  float* spdf  = allocF((size_t)cE * 128);
  float* gatef = allocF((size_t)cE * 128);
  float* eaF   = allocF((size_t)cE * 128);
  float* eatmp = allocF((size_t)cE * 128);
  float* eacmb = allocF((size_t)cE * 128);
  float* eh[4]; for (int l = 0; l < 4; ++l) eh[l] = allocF((size_t)cE * 128);
  float* cat2  = allocF((size_t)cE * 256);
  float* ee    = allocF((size_t)cE * 256);
  float* raw   = allocF((size_t)cE * 256);
  float* cat6  = allocF((size_t)cE * 640);
  float* S8    = allocF((size_t)cE * 8);
  float* mx8   = allocF((size_t)cNT * 8);
  float* den8  = allocF((size_t)cNT * 8);
  int*   deg   = allocI((size_t)cNT);
  int*   dm    = allocI((size_t)cB * 16384);
  float* biasT = allocF((size_t)cB * 16384 * 8);
  float* cat5  = allocF((size_t)cNT * 512);
  float* hnode = allocF((size_t)cNT * 128);
  float* qd    = allocF((size_t)cNT * 128);
  float* kd    = allocF((size_t)cE * 128);
  float* vd    = allocF((size_t)cE * 128);
  float* Sx    = allocF((size_t)cE * 4);
  float* mx4   = allocF((size_t)cNT * 4);
  float* den4  = allocF((size_t)cNT * 4);
  float* eaggi = allocF((size_t)cNT * 128);
  float* eagg  = allocF((size_t)cNT * 128);
  float* meanp = allocF((size_t)cB * 256);
  float* maxp  = allocF((size_t)cB * 256);
  float* catp  = allocF((size_t)cB * 512);
  float* gpool = allocF((size_t)cB * 128);
  float* emean = allocF((size_t)cB * 128);
  float* gedge = allocF((size_t)cB * 128);
  float* xf    = allocF((size_t)cNT * 512);
  float* h1o   = allocF((size_t)cNT * 256);
  float* h2o   = allocF((size_t)cNT * 128);

  auto GEMM = [&](const float* A, int wi, int bi, float* C, int M, int K, int N, int act) {
    int ntiles = (N + 15) >> 4;
    int npk = (K >> 5) * ntiles * 512;
    k_pack_w<<<G1(npk), dim3(256), 0, stream>>>(F(wi), Wbf, K, N, ntiles);
    dim3 g((N + 127) / 128, (M + 63) / 64);
    if (act == 0)      k_gemm<0><<<g, dim3(256), 0, stream>>>(A, Wbf, F(bi), C, M, K, N);
    else if (act == 1) k_gemm<1><<<g, dim3(256), 0, stream>>>(A, Wbf, F(bi), C, M, K, N);
    else if (act == 2) k_gemm<2><<<g, dim3(256), 0, stream>>>(A, Wbf, F(bi), C, M, K, N);
    else               k_gemm<3><<<g, dim3(256), 0, stream>>>(A, Wbf, F(bi), C, M, K, N);
  };

  // ---- input projections ----
  GEMM(in_x, 132, 131, bufA, cNT, 128, 256, 1);
  k_bn<<<G1((long)cNT * 256), dim3(256), 0, stream>>>(bufA, x0, F(128), F(127), F(129), F(130), (long)cNT * 256, 256);
  k_copy<<<G1((long)cNT * 256), dim3(256), 0, stream>>>(xcur, x0, (long)cNT * 256);

  GEMM(in_ea, 15, 14, eatmp, cE, 64, 128, 1);
  k_bn<<<G1((long)cE * 128), dim3(256), 0, stream>>>(eatmp, ea_hc, F(11), F(10), F(12), F(13), (long)cE * 128, 128);
  k_gather_emb<<<G1((long)cE * 128), dim3(256), 0, stream>>>(spd, F(145), spdf, cE, 128, 32);
  k_concat2<<<G1((long)cE * 256), dim3(256), 0, stream>>>(ea_hc, 128, spdf, 128, cat2, cE);
  GEMM(cat2, 7, 6, gatef, cE, 256, 128, 3);
  k_fuse<<<G1((long)cE * 128), dim3(256), 0, stream>>>(gatef, ea_hc, spdf, eatmp, (long)cE * 128);
  k_ln<<<dim3((cE + 7) / 8), dim3(256), 0, stream>>>(eatmp, eaF, F(9), F(8), cE, 128);

  // ---- layers ----
  for (int l = 0; l < 4; ++l) {
    int L = 35 + 23 * l;
    k_fill_i<<<G1(cNT), dim3(256), 0, stream>>>(deg, 0, cNT);
    k_deg<<<G1(cE), dim3(256), 0, stream>>>(src, deg, cE);
    k_add_gather<<<G1((long)cNT * 256), dim3(256), 0, stream>>>(xcur, deg, F(L + 0), bufA, cNT, 256, 64);
    k_ln<<<dim3((cNT + 7) / 8), dim3(256), 0, stream>>>(bufA, bufH, F(L + 16), F(L + 15), cNT, 256);
    GEMM(bufH, L + 18, L + 17, qb, cNT, 256, 256, 0);
    GEMM(bufH, L + 14, L + 13, kb, cNT, 256, 256, 0);
    GEMM(bufH, L + 22, L + 21, vb, cNT, 256, 256, 0);
    GEMM(bufH, L + 20, L + 19, skp, cNT, 256, 256, 0);
    GEMM(eaF, L + 2, L + 1, ee, cE, 128, 256, 0);
    k_edge_score<<<G1((long)cE * 8), dim3(256), 0, stream>>>(qb, kb, ee, src, dst, S8, cE);
    k_fill_f<<<G1((long)cNT * 8), dim3(256), 0, stream>>>(mx8, -1e30f, (long)cNT * 8);
    k_fill_f<<<G1((long)cNT * 8), dim3(256), 0, stream>>>(den8, 0.f, (long)cNT * 8);
    k_seg_max<<<G1((long)cE * 8), dim3(256), 0, stream>>>(S8, dst, mx8, cE, 8);
    k_seg_expsum<<<G1((long)cE * 8), dim3(256), 0, stream>>>(S8, dst, mx8, den8, cE, 8);
    k_seg_norm<<<G1((long)cE * 8), dim3(256), 0, stream>>>(S8, dst, den8, cE, 8);
    k_fill_f<<<G1((long)cNT * 256), dim3(256), 0, stream>>>(agg, 0.f, (long)cNT * 256);
    k_edge_agg<<<G1((long)cE * 256), dim3(256), 0, stream>>>(vb, ee, S8, src, dst, agg, cE);
    k_add3<<<G1((long)cNT * 256), dim3(256), 0, stream>>>(bufA, agg, skp, xat, (long)cNT * 256);
    k_ln<<<dim3((cNT + 7) / 8), dim3(256), 0, stream>>>(xat, bufH, F(L + 16), F(L + 15), cNT, 256);
    GEMM(bufH, L + 10, L + 9, ffn_t, cNT, 256, 1024, 2);
    GEMM(ffn_t, L + 12, L + 11, ffn_o, cNT, 1024, 256, 0);
    k_add2<<<G1((long)cNT * 256), dim3(256), 0, stream>>>(xat, ffn_o, xcur, (long)cNT * 256);
    k_ln<<<dim3((cNT + 7) / 8), dim3(256), 0, stream>>>(xcur, xnb, F(L + 8), F(L + 7), cNT, 256);
    k_eu_concat<<<G1((long)cE * 640), dim3(256), 0, stream>>>(xnb, src, dst, eaF, cat6, cE);
    GEMM(cat6, L + 6, L + 5, raw, cE, 640, 256, 1);
    k_eu_update<<<G1((long)cE * 128), dim3(256), 0, stream>>>(eaF, raw, eatmp, cE);
    k_ln<<<dim3((cE + 7) / 8), dim3(256), 0, stream>>>(eatmp, eaF, F(L + 4), F(L + 3), cE, 128);
    k_copy<<<G1((long)cNT * 256), dim3(256), 0, stream>>>(xh[l], xcur, (long)cNT * 256);
    k_copy<<<G1((long)cE * 128), dim3(256), 0, stream>>>(eh[l], eaF, (long)cE * 128);
  }

  k_add3<<<G1((long)cNT * 256), dim3(256), 0, stream>>>(xh[1], xh[2], xh[3], xcomb, (long)cNT * 256);
  k_add3<<<G1((long)cE * 128), dim3(256), 0, stream>>>(eh[1], eh[2], eh[3], eacmb, (long)cE * 128);

  // ---- global transformer ----
  k_fill_i<<<G1((long)cB * 16384), dim3(256), 0, stream>>>(dm, 0, (long)cB * 16384);
  k_dm<<<G1(cE), dim3(256), 0, stream>>>(src, dst, spd, dm, cE);
  k_bias<<<G1((long)cB * 16384), dim3(256), 0, stream>>>(pos, dm, F(16), F(26), F(25), biasT, cB * 16384);
  k_ln<<<dim3((cNT + 7) / 8), dim3(256), 0, stream>>>(xcomb, hN, F(22), F(21), cNT, 256);
  GEMM(hN, 32, 31, qg, cNT, 256, 256, 0);
  GEMM(hN, 28, 27, kg, cNT, 256, 256, 0);
  GEMM(hN, 34, 33, vg, cNT, 256, 256, 0);
  k_gattn<<<dim3(cB * 8), dim3(256), 0, stream>>>(qg, kg, vg, biasT, ao);
  GEMM(ao, 30, 29, ffn_o, cNT, 256, 256, 0);
  k_add2<<<G1((long)cNT * 256), dim3(256), 0, stream>>>(hN, ffn_o, hN, (long)cNT * 256);
  k_ln<<<dim3((cNT + 7) / 8), dim3(256), 0, stream>>>(hN, bufH, F(24), F(23), cNT, 256);
  GEMM(bufH, 18, 17, ffn_t, cNT, 256, 1024, 2);
  GEMM(ffn_t, 20, 19, ffn_o, cNT, 1024, 256, 0);
  k_add2<<<G1((long)cNT * 256), dim3(256), 0, stream>>>(hN, ffn_o, hN, (long)cNT * 256);
  k_add2<<<G1((long)cNT * 256), dim3(256), 0, stream>>>(xcomb, hN, xcomb, (long)cNT * 256);

  // ---- readout ----
  k_concat2<<<G1((long)cNT * 512), dim3(256), 0, stream>>>(xcomb, 256, x0, 256, cat5, cNT);
  GEMM(cat5, 142, 141, hnode, cNT, 512, 128, 1);
  GEMM(hnode, 151, 150, qd, cNT, 128, 128, 0);
  GEMM(eacmb, 147, 146, kd, cE, 128, 128, 0);
  GEMM(eacmb, 153, 152, vd, cE, 128, 128, 0);
  k_x_score<<<G1((long)cE * 4), dim3(256), 0, stream>>>(qd, kd, src, Sx, cE);
  k_fill_f<<<G1((long)cNT * 4), dim3(256), 0, stream>>>(mx4, -1e30f, (long)cNT * 4);
  k_fill_f<<<G1((long)cNT * 4), dim3(256), 0, stream>>>(den4, 0.f, (long)cNT * 4);
  k_seg_max<<<G1((long)cE * 4), dim3(256), 0, stream>>>(Sx, src, mx4, cE, 4);
  k_seg_expsum<<<G1((long)cE * 4), dim3(256), 0, stream>>>(Sx, src, mx4, den4, cE, 4);
  k_seg_norm<<<G1((long)cE * 4), dim3(256), 0, stream>>>(Sx, src, den4, cE, 4);
  k_fill_f<<<G1((long)cNT * 128), dim3(256), 0, stream>>>(eaggi, 0.f, (long)cNT * 128);
  k_x_agg<<<G1((long)cE * 128), dim3(256), 0, stream>>>(vd, Sx, src, eaggi, cE);
  GEMM(eaggi, 149, 148, eagg, cNT, 128, 128, 0);
  k_pool<<<G1((long)cB * 256), dim3(256), 0, stream>>>(xcomb, meanp, maxp);
  k_concat2<<<G1((long)cB * 512), dim3(256), 0, stream>>>(meanp, 256, maxp, 256, catp, cB);
  GEMM(catp, 144, 143, gpool, cB, 512, 128, 1);
  k_fill_f<<<G1((long)cB * 128), dim3(256), 0, stream>>>(emean, 0.f, (long)cB * 128);
  k_emean<<<G1((long)cE * 128), dim3(256), 0, stream>>>(eacmb, src, emean, cE);
  GEMM(emean, 134, 133, gedge, cB, 128, 128, 1);
  k_cat4<<<G1((long)cNT * 512), dim3(256), 0, stream>>>(hnode, eagg, gpool, gedge, batch, xf, cNT);
  GEMM(xf, 136, 135, h1o, cNT, 512, 256, 1);
  GEMM(h1o, 138, 137, h2o, cNT, 256, 128, 1);
  GEMM(h2o, 140, 139, (float*)d_out, cNT, 128, 8, 0);
}
